// GNNModel_24386824306776
// MI455X (gfx1250) — matrix operand order repaired and checked by reading, not compiled, not run on hardware
//
#include <hip/hip_runtime.h>

typedef __attribute__((ext_vector_type(16))) _Float16 v16h;
typedef __attribute__((ext_vector_type(8)))  _Float16 v8h;
typedef __attribute__((ext_vector_type(8)))  float    v8f;
typedef __attribute__((ext_vector_type(4)))  float    v4f;

#define NNODES 100000
#define NEDGES 1600000
#define HDIM   128
#define SCORER_WAVES 8

static __device__ __forceinline__ v8f wmma16(v16h a, v16h b, v8f c) {
  return __builtin_amdgcn_wmma_f32_16x16x32_f16(false, a, false, b, (short)0, c, false, false);
}

// ---------------- utility kernels ----------------

__global__ void k_izero(int* __restrict__ p, int n) {
  int i = blockIdx.x * blockDim.x + threadIdx.x;
  if (i < n) p[i] = 0;
}

__global__ void k_degc(const int* __restrict__ tgt, int* __restrict__ degc, int E) {
  int e = blockIdx.x * blockDim.x + threadIdx.x;
  if (e < E) atomicAdd(&degc[tgt[e]], 1);
}

__global__ void k_dis(const int* __restrict__ degc, float* __restrict__ dis, int n) {
  int i = blockIdx.x * blockDim.x + threadIdx.x;
  if (i < n) dis[i] = rsqrtf((float)degc[i] + 1.0f);
}

// ---- exclusive scan of degc -> offs (CSR row offsets) ----

__global__ void k_scan_block(const int* __restrict__ in, int* __restrict__ out,
                             int* __restrict__ bsum, int n) {
  __shared__ int tmp[256];
  int i = blockIdx.x * 256 + threadIdx.x;
  int v = (i < n) ? in[i] : 0;
  tmp[threadIdx.x] = v;
  __syncthreads();
#pragma unroll
  for (int ofs = 1; ofs < 256; ofs <<= 1) {
    int t = (threadIdx.x >= ofs) ? tmp[threadIdx.x - ofs] : 0;
    __syncthreads();
    tmp[threadIdx.x] += t;
    __syncthreads();
  }
  if (i < n) out[i] = tmp[threadIdx.x] - v;   // exclusive
  if (threadIdx.x == 255) bsum[blockIdx.x] = tmp[255];
}

__global__ void k_scan_bsum(int* __restrict__ bsum, int nb) {
  if (blockIdx.x == 0 && threadIdx.x == 0) {
    int acc = 0;
    for (int i = 0; i < nb; ++i) { int v = bsum[i]; bsum[i] = acc; acc += v; }
  }
}

__global__ void k_scan_add(int* __restrict__ out, const int* __restrict__ bsum, int n) {
  int i = blockIdx.x * 256 + threadIdx.x;
  if (i < n) out[i] += bsum[blockIdx.x];
}

__global__ void k_fill(const int* __restrict__ tgt, const int* __restrict__ offs,
                       int* __restrict__ cursor, int* __restrict__ eid, int E) {
  int e = blockIdx.x * blockDim.x + threadIdx.x;
  if (e < E) {
    int t = tgt[e];
    int p = atomicAdd(&cursor[t], 1);
    eid[offs[t] + p] = e;
  }
}

// Pack f32 weight [K][N] row-major into f16 WMMA B fragments.
//   lane<16 : n = nt*16+lane,    k = kt*32 + (h<8 ? h : 16 + h-8)
//   lane>=16: n = nt*16+lane-16, k = kt*32 + 8 + (h<8 ? h : 16 + h-8)
__global__ void k_pack_b(const float* __restrict__ W, _Float16* __restrict__ out,
                         int K, int N) {
  int tid = blockIdx.x * blockDim.x + threadIdx.x;
  if (tid >= K * N) return;
  int f    = tid >> 9;
  int r    = tid & 511;
  int lane = r >> 4;
  int h    = r & 15;
  int ntiles = N >> 4;
  int nt = f % ntiles;
  int kt = f / ntiles;
  int n = nt * 16 + (lane & 15);
  int k = kt * 32 + ((lane >> 4) << 3) + (h & 7) + ((h >> 3) << 4);
  out[tid] = (_Float16)W[(size_t)k * N + n];
}

// ---------------- GCN dense: hw = h @ W ----------------
__global__ void __launch_bounds__(32)
k_gemm_hw(const float* __restrict__ hin, const _Float16* __restrict__ Wp,
          float* __restrict__ hw) {
  int lane = threadIdx.x;
  int mT = blockIdx.x, nT = blockIdx.y;
  int m    = mT * 16 + (lane & 15);
  int koff = (lane >> 4) << 3;   // 0 or 8

  v8f acc = {};
#pragma unroll
  for (int kt = 0; kt < 4; ++kt) {
    const float* bp = hin + (size_t)m * HDIM + kt * 32 + koff;
    v4f f0 = *(const v4f*)(bp);
    v4f f1 = *(const v4f*)(bp + 4);
    v4f f2 = *(const v4f*)(bp + 16);
    v4f f3 = *(const v4f*)(bp + 20);
    v16h a;
#pragma unroll
    for (int i = 0; i < 4; ++i) {
      a[i]      = (_Float16)f0[i];
      a[4 + i]  = (_Float16)f1[i];
      a[8 + i]  = (_Float16)f2[i];
      a[12 + i] = (_Float16)f3[i];
    }
    v16h b = *(const v16h*)(Wp + ((size_t)(kt * 8 + nT) * 512 + lane * 16));
    acc = wmma16(a, b, acc);
  }
  int ncol  = nT * 16 + (lane & 15);
  int mbase = mT * 16 + ((lane >> 4) << 3);
#pragma unroll
  for (int r = 0; r < 8; ++r)
    hw[(size_t)(mbase + r) * HDIM + ncol] = acc[r];
}

// ---------------- CSR aggregation (no atomics): one block per node ----------------
__global__ void __launch_bounds__(128)
k_gather_agg(const float* __restrict__ hw, const int* __restrict__ src,
             const int* __restrict__ eid, const int* __restrict__ offs,
             const int* __restrict__ degc, const float* __restrict__ dis,
             const float* __restrict__ bias, float* __restrict__ hout) {
  int i = blockIdx.x;
  int d = threadIdx.x;
  int beg = offs[i];
  int cnt = degc[i];
  float di = dis[i];
  float acc = 0.0f;
  for (int j = 0; j < cnt; ++j) {
    int e = eid[beg + j];
    int s = src[e];
    acc += hw[(size_t)s * HDIM + d] * (dis[s] * di);
  }
  float v = acc + hw[(size_t)i * HDIM + d] * di * di + bias[d];
  hout[(size_t)i * HDIM + d] = fmaxf(v, 0.0f);
}

__global__ void k_embout(const float* __restrict__ h, float* __restrict__ out_f32,
                         _Float16* __restrict__ emb16) {
  long long tid = (long long)blockIdx.x * blockDim.x + threadIdx.x;
  if (tid >= (long long)NNODES * HDIM) return;
  float v = h[tid];
  if (v != v) v = 0.0f;
  out_f32[tid] = v;
  emb16[tid] = (_Float16)v;
}

// ---------------- edge scorer MLP ----------------
// 8 waves/block; double-buffered LDS staging of B fragments so global loads
// for slice kt+1 overlap the wmma stream of slice kt.

__global__ void __launch_bounds__(256)
k_scorer(const _Float16* __restrict__ emb, const int* __restrict__ src,
         const int* __restrict__ tgt,
         const _Float16* __restrict__ W1p, const float* __restrict__ b1,
         const _Float16* __restrict__ W2p, const float* __restrict__ b2,
         const float* __restrict__ W3,   const float* __restrict__ b3,
         float* __restrict__ score) {
  __shared__ __align__(16) _Float16 bstage[2][16 * 512];                 // 2 x 16 KB
  __shared__ __align__(16) _Float16 zbuf[SCORER_WAVES][16 * 256];        // 8 x 8 KB
  int tid  = threadIdx.x;
  int lane = tid & 31;
  int w    = tid >> 5;
  int e0   = (blockIdx.x * SCORER_WAVES + w) * 16;
  int erow = e0 + (lane & 15);
  int s = src[erow], t = tgt[erow];
  int koff  = (lane >> 4) << 3;   // 0 or 8
  int ncol0 = lane & 15;
  int mb    = (lane >> 4) << 3;

  const v8f vz = {};

  // ======== layer 1: [16 x 256] @ [256 x 256] ========
  v8f acc[16];
#pragma unroll
  for (int i = 0; i < 16; ++i) acc[i] = vz;

  // prologue: stage slice 0
  {
    const v8h* sp = (const v8h*)W1p;
    v8h* dp = (v8h*)bstage[0];
#pragma unroll
    for (int c = 0; c < 4; ++c) dp[tid + c * 256] = sp[tid + c * 256];
  }

  for (int kt = 0; kt < 8; ++kt) {
    __syncthreads();                      // staged slice kt now visible
    int cur = kt & 1;

    // issue global loads for slice kt+1 (results consumed after wmma stream)
    v8h pre[4];
    if (kt < 7) {
      const v8h* sp = (const v8h*)(W1p + (size_t)(kt + 1) * 16 * 512);
#pragma unroll
      for (int c = 0; c < 4; ++c) pre[c] = sp[tid + c * 256];
    }

    // A fragment: gathered edge features (src emb for kt<4, tgt emb for kt>=4)
    const _Float16* rb = (kt < 4) ? (emb + (size_t)s * HDIM + kt * 32)
                                  : (emb + (size_t)t * HDIM + (kt - 4) * 32);
    v8h c0 = *(const v8h*)(rb + koff);
    v8h c1 = *(const v8h*)(rb + 16 + koff);
    v16h a;
#pragma unroll
    for (int i = 0; i < 8; ++i) { a[i] = c0[i]; a[8 + i] = c1[i]; }

#pragma unroll
    for (int nt = 0; nt < 16; ++nt) {
      v16h b = *(const v16h*)(bstage[cur] + nt * 512 + lane * 16);
      acc[nt] = wmma16(a, b, acc[nt]);
    }

    // write next slice into the other buffer
    if (kt < 7) {
      v8h* dp = (v8h*)bstage[cur ^ 1];
#pragma unroll
      for (int c = 0; c < 4; ++c) dp[tid + c * 256] = pre[c];
    }
  }

#pragma unroll
  for (int nt = 0; nt < 16; ++nt) {
    int n = nt * 16 + ncol0;
    float bi = b1[n];
#pragma unroll
    for (int r = 0; r < 8; ++r)
      zbuf[w][(mb + r) * 256 + n] = (_Float16)fmaxf(acc[nt][r] + bi, 0.0f);
  }

  // ======== layer 2: [16 x 256] @ [256 x 128] ========
  v8f acc2[8];
#pragma unroll
  for (int i = 0; i < 8; ++i) acc2[i] = vz;

  // prologue: stage slice 0 (safe: all waves are past the last layer-1 stage)
  {
    const v8h* sp = (const v8h*)W2p;
    v8h* dp = (v8h*)bstage[0];
#pragma unroll
    for (int c = 0; c < 2; ++c) dp[tid + c * 256] = sp[tid + c * 256];
  }

  for (int kt = 0; kt < 8; ++kt) {
    __syncthreads();
    int cur = kt & 1;

    v8h pre[2];
    if (kt < 7) {
      const v8h* sp = (const v8h*)(W2p + (size_t)(kt + 1) * 8 * 512);
#pragma unroll
      for (int c = 0; c < 2; ++c) pre[c] = sp[tid + c * 256];
    }

    const _Float16* rb = zbuf[w] + (size_t)(lane & 15) * 256 + kt * 32;
    v8h c0 = *(const v8h*)(rb + koff);
    v8h c1 = *(const v8h*)(rb + 16 + koff);
    v16h a;
#pragma unroll
    for (int i = 0; i < 8; ++i) { a[i] = c0[i]; a[8 + i] = c1[i]; }

#pragma unroll
    for (int nt = 0; nt < 8; ++nt) {
      v16h b = *(const v16h*)(bstage[cur] + nt * 512 + lane * 16);
      acc2[nt] = wmma16(a, b, acc2[nt]);
    }

    if (kt < 7) {
      v8h* dp = (v8h*)bstage[cur ^ 1];
#pragma unroll
      for (int c = 0; c < 2; ++c) dp[tid + c * 256] = pre[c];
    }
  }

  // z1 fully consumed by this wave (LDS ops are in-order per wave) -> reuse zbuf
#pragma unroll
  for (int nt = 0; nt < 8; ++nt) {
    int n = nt * 16 + ncol0;
    float bi = b2[n];
#pragma unroll
    for (int r = 0; r < 8; ++r)
      zbuf[w][(mb + r) * 128 + n] = (_Float16)fmaxf(acc2[nt][r] + bi, 0.0f);
  }

  // ======== layer 3: [16 x 128] @ [128 x 1] ========
  if (lane < 16) {
    float sc = b3[0];
#pragma unroll 8
    for (int k = 0; k < 128; ++k)
      sc += (float)zbuf[w][lane * 128 + k] * W3[k];
    sc = fminf(fmaxf(sc, -1e6f), 1e6f);
    if (sc != sc) sc = 0.0f;
    score[e0 + lane] = sc;
  }
}

// ---------------- host launch ----------------

extern "C" void kernel_launch(void* const* d_in, const int* in_sizes, int n_in,
                              void* d_out, int out_size, void* d_ws, size_t ws_size,
                              hipStream_t stream) {
  const float* x   = (const float*)d_in[0];
  const int*   ei  = (const int*)d_in[1];
  const float* W1  = (const float*)d_in[2];
  const float* b1  = (const float*)d_in[3];
  const float* W2  = (const float*)d_in[4];
  const float* b2  = (const float*)d_in[5];
  const float* W3  = (const float*)d_in[6];
  const float* b3  = (const float*)d_in[7];
  const float* sW1 = (const float*)d_in[8];
  const float* sb1 = (const float*)d_in[9];
  const float* sW2 = (const float*)d_in[10];
  const float* sb2 = (const float*)d_in[11];
  const float* sW3 = (const float*)d_in[12];
  const float* sb3 = (const float*)d_in[13];
  const int* src = ei;
  const int* tgt = ei + NEDGES;

  char* ws = (char*)d_ws;
  size_t off = 0;
  auto alloc = [&](size_t bytes) -> void* {
    void* p = ws + off;
    off += (bytes + 255) & ~(size_t)255;
    return p;
  };

  int*       degc   = (int*)alloc((size_t)NNODES * 4);
  int*       offs   = (int*)alloc((size_t)NNODES * 4);
  int*       cursor = (int*)alloc((size_t)NNODES * 4);
  int*       bsum   = (int*)alloc((size_t)512 * 4);
  int*       eid    = (int*)alloc((size_t)NEDGES * 4);
  float*     dis    = (float*)alloc((size_t)NNODES * 4);
  float*     bufA   = (float*)alloc((size_t)NNODES * HDIM * 4);
  float*     bufB   = (float*)alloc((size_t)NNODES * HDIM * 4);
  float*     bufC   = (float*)alloc((size_t)NNODES * HDIM * 4);
  _Float16*  emb16  = (_Float16*)alloc((size_t)NNODES * HDIM * 2);
  _Float16*  W1p    = (_Float16*)alloc((size_t)128 * 128 * 2);
  _Float16*  W2p    = (_Float16*)alloc((size_t)128 * 128 * 2);
  _Float16*  W3p    = (_Float16*)alloc((size_t)128 * 128 * 2);
  _Float16*  sW1p   = (_Float16*)alloc((size_t)256 * 256 * 2);
  _Float16*  sW2p   = (_Float16*)alloc((size_t)256 * 128 * 2);

  const long long NH = (long long)NNODES * HDIM;
  const int nblk = (NNODES + 255) / 256;   // 391

  // ---- degree, normalization, CSR build ----
  k_izero<<<nblk, 256, 0, stream>>>(degc, NNODES);
  k_izero<<<nblk, 256, 0, stream>>>(cursor, NNODES);
  k_degc<<<(NEDGES + 255) / 256, 256, 0, stream>>>(tgt, degc, NEDGES);
  k_dis<<<nblk, 256, 0, stream>>>(degc, dis, NNODES);
  k_scan_block<<<nblk, 256, 0, stream>>>(degc, offs, bsum, NNODES);
  k_scan_bsum<<<1, 32, 0, stream>>>(bsum, nblk);
  k_scan_add<<<nblk, 256, 0, stream>>>(offs, bsum, NNODES);
  k_fill<<<(NEDGES + 255) / 256, 256, 0, stream>>>(tgt, offs, cursor, eid, NEDGES);

  // ---- weight packing (f32 -> f16 B fragments) ----
  k_pack_b<<<(128 * 128 + 255) / 256, 256, 0, stream>>>(W1, W1p, 128, 128);
  k_pack_b<<<(128 * 128 + 255) / 256, 256, 0, stream>>>(W2, W2p, 128, 128);
  k_pack_b<<<(128 * 128 + 255) / 256, 256, 0, stream>>>(W3, W3p, 128, 128);
  k_pack_b<<<(256 * 256 + 255) / 256, 256, 0, stream>>>(sW1, sW1p, 256, 256);
  k_pack_b<<<(256 * 128 + 255) / 256, 256, 0, stream>>>(sW2, sW2p, 256, 128);

  dim3 ggrid(NNODES / 16, HDIM / 16);

  // layer 1: in = x,   hw = bufA, out = bufB
  k_gemm_hw<<<ggrid, 32, 0, stream>>>(x, W1p, bufA);
  k_gather_agg<<<NNODES, 128, 0, stream>>>(bufA, src, eid, offs, degc, dis, b1, bufB);
  // layer 2: in = bufB, hw = bufA, out = bufC
  k_gemm_hw<<<ggrid, 32, 0, stream>>>(bufB, W2p, bufA);
  k_gather_agg<<<NNODES, 128, 0, stream>>>(bufA, src, eid, offs, degc, dis, b2, bufC);
  // layer 3: in = bufC, hw = bufA, out = bufB
  k_gemm_hw<<<ggrid, 32, 0, stream>>>(bufC, W3p, bufA);
  k_gather_agg<<<NNODES, 128, 0, stream>>>(bufA, src, eid, offs, degc, dis, b3, bufB);

  // outputs: d_out = [score (E) | node_embs (N*H)]
  float* out_score = (float*)d_out;
  float* out_emb   = out_score + NEDGES;
  k_embout<<<(unsigned)((NH + 255) / 256), 256, 0, stream>>>(bufB, out_emb, emb16);

  // edge scorer MLP (WMMA-heavy, double-buffered LDS-staged weights)
  k_scorer<<<NEDGES / (16 * SCORER_WAVES), 256, 0, stream>>>(
      emb16, src, tgt, sW1p, sb1, sW2p, sb2, sW3, sb3, out_score);
}
